// HomogeneousGraphConvolution_74028056314526
// MI455X (gfx1250) — matrix operand order repaired and checked by reading, not compiled, not run on hardware
//
#include <hip/hip_runtime.h>
#include <hip/hip_bf16.h>

typedef __attribute__((ext_vector_type(16))) __bf16 v16bf;
typedef __attribute__((ext_vector_type(8)))  float  v8f;

#define DIM 128

// ---------------------------------------------------------------------------
// Zero the accumulator region (summed[N*128] + deg[N]) every call.
// ---------------------------------------------------------------------------
__global__ void zero_kernel(float4* __restrict__ p, int n4) {
  int i = blockIdx.x * blockDim.x + threadIdx.x;
  if (i < n4) p[i] = make_float4(0.f, 0.f, 0.f, 0.f);
}

// ---------------------------------------------------------------------------
// Swizzle a 128x128 f32 weight (row-major, W[k][n]) into the CDNA5 WMMA
// bf16 B-operand layout: for K-tile kt (32 rows of K) and N-tile nt (16 cols),
// lane L holds column n = nt*16 + (L&15) and K values (L>>4)*8 + e for e<8,
// 16 + (L>>4)*8 + (e-8) for e>=8, stored as 16 contiguous bf16 per lane.
// ---------------------------------------------------------------------------
__global__ void swizzle_w_kernel(const float* __restrict__ W,
                                 __hip_bfloat16* __restrict__ Wswz) {
  int t = blockIdx.x * blockDim.x + threadIdx.x;   // 0 .. 16383
  if (t >= DIM * DIM) return;
  int elem = t & 15;
  int L    = (t >> 4) & 31;
  int nt   = (t >> 9) & 7;
  int kt   = t >> 12;
  int half = L >> 4;
  int n = nt * 16 + (L & 15);
  int k = kt * 32 + (elem < 8 ? half * 8 + elem : 16 + half * 8 + (elem - 8));
  float v = W[k * DIM + n];
  Wswz[t] = __hip_bfloat16(v);
}

// ---------------------------------------------------------------------------
// Scatter-add: one wave32 per edge. Each lane gathers a float4 of x[src]
// and atomically accumulates into summed[dst]; lane 0 bumps deg[dst].
// ---------------------------------------------------------------------------
__global__ __launch_bounds__(256)
void scatter_add_kernel(const float* __restrict__ x,
                        const int* __restrict__ ei,
                        float* __restrict__ summed,
                        float* __restrict__ deg, int E) {
  unsigned gtid = blockIdx.x * 256u + threadIdx.x;
  int e    = (int)(gtid >> 5);
  int lane = threadIdx.x & 31;
  if (e >= E) return;
  int s = ei[e];
  int d = ei[E + e];
  const float4 v = *(const float4*)(x + (size_t)s * DIM + lane * 4);
  float* base = summed + (size_t)d * DIM + lane * 4;
  atomicAdd(base + 0, v.x);
  atomicAdd(base + 1, v.y);
  atomicAdd(base + 2, v.z);
  atomicAdd(base + 3, v.w);
  if (lane == 0) atomicAdd(&deg[d], 1.0f);
}

// ---------------------------------------------------------------------------
// Fused: h = mean @ W_l + b_l + x @ W_r  -> LayerNorm -> exact GELU.
// One block (8 waves) per 16-row tile; wave w owns output columns [w*16,w*16+16).
// 8 x v_wmma_f32_16x16x32_bf16 per wave (4 K-tiles x 2 matrices).
// ---------------------------------------------------------------------------
__device__ __forceinline__ void pack8(v16bf& v, int off, float4 a, float4 b) {
  v[off + 0] = (__bf16)a.x; v[off + 1] = (__bf16)a.y;
  v[off + 2] = (__bf16)a.z; v[off + 3] = (__bf16)a.w;
  v[off + 4] = (__bf16)b.x; v[off + 5] = (__bf16)b.y;
  v[off + 6] = (__bf16)b.z; v[off + 7] = (__bf16)b.w;
}

__device__ __forceinline__ float gelu_exact(float t) {
  return 0.5f * t * (1.0f + erff(t * 0.70710678118654752f));
}

__global__ __launch_bounds__(256)
void gemm_ln_gelu_kernel(const float* __restrict__ x,
                         const float* __restrict__ summed,
                         const float* __restrict__ deg,
                         const void* __restrict__ wl_swz_,
                         const void* __restrict__ wr_swz_,
                         const float* __restrict__ bias,
                         const float* __restrict__ gamma,
                         const float* __restrict__ beta,
                         float* __restrict__ out) {
  __shared__ float hbuf[16][DIM + 4];

  const v16bf* wl_swz = (const v16bf*)wl_swz_;
  const v16bf* wr_swz = (const v16bf*)wr_swz_;

  const int wave = threadIdx.x >> 5;   // 0..7 == output N-tile
  const int lane = threadIdx.x & 31;
  const int half = lane >> 4;
  const int l15  = lane & 15;

  const int row = blockIdx.x * 16 + l15;          // A-operand row for this lane
  const float invdeg = 1.0f / fmaxf(deg[row], 1.0f);

  const float* xr = x      + (size_t)row * DIM;
  const float* sr = summed + (size_t)row * DIM;

  v8f c = {};
#pragma unroll
  for (int kt = 0; kt < 4; ++kt) {
    const int k0 = kt * 32 + half * 8;
    // A from x (f32 -> bf16 in-register)
    float4 x0 = *(const float4*)(xr + k0);
    float4 x1 = *(const float4*)(xr + k0 + 4);
    float4 x2 = *(const float4*)(xr + k0 + 16);
    float4 x3 = *(const float4*)(xr + k0 + 20);
    // A from mean = summed * invdeg
    float4 s0 = *(const float4*)(sr + k0);
    float4 s1 = *(const float4*)(sr + k0 + 4);
    float4 s2 = *(const float4*)(sr + k0 + 16);
    float4 s3 = *(const float4*)(sr + k0 + 20);
    s0.x *= invdeg; s0.y *= invdeg; s0.z *= invdeg; s0.w *= invdeg;
    s1.x *= invdeg; s1.y *= invdeg; s1.z *= invdeg; s1.w *= invdeg;
    s2.x *= invdeg; s2.y *= invdeg; s2.z *= invdeg; s2.w *= invdeg;
    s3.x *= invdeg; s3.y *= invdeg; s3.z *= invdeg; s3.w *= invdeg;

    v16bf ax, am;
    pack8(ax, 0, x0, x1); pack8(ax, 8, x2, x3);
    pack8(am, 0, s0, s1); pack8(am, 8, s2, s3);

    v16bf bl = wl_swz[(kt * 8 + wave) * 32 + lane];
    v16bf br = wr_swz[(kt * 8 + wave) * 32 + lane];

    c = __builtin_amdgcn_wmma_f32_16x16x32_bf16(false, am, false, bl,
                                                (short)0, c, false, false);
    c = __builtin_amdgcn_wmma_f32_16x16x32_bf16(false, ax, false, br,
                                                (short)0, c, false, false);
  }

  // D layout: lane column = l15, VGPR r -> local row half*8 + r.
  const int col = wave * 16 + l15;
  const float bv = bias[col];
#pragma unroll
  for (int r = 0; r < 8; ++r)
    hbuf[half * 8 + r][col] = c[r] + bv;

  __syncthreads();

  // LayerNorm + GELU: each wave handles 2 of the 16 rows; 4 features/lane.
  const float4 g  = *(const float4*)(gamma + lane * 4);
  const float4 bb = *(const float4*)(beta  + lane * 4);
#pragma unroll
  for (int i = 0; i < 2; ++i) {
    const int rr = wave * 2 + i;
    float4 v = *(const float4*)(&hbuf[rr][lane * 4]);
    float s = v.x + v.y + v.z + v.w;
    float q = v.x * v.x + v.y * v.y + v.z * v.z + v.w * v.w;
#pragma unroll
    for (int m = 16; m > 0; m >>= 1) {
      s += __shfl_xor(s, m, 32);
      q += __shfl_xor(q, m, 32);
    }
    const float mu  = s * (1.0f / 128.0f);
    const float var = q * (1.0f / 128.0f) - mu * mu;
    const float rs  = rsqrtf(var + 1e-5f);
    float4 o;
    o.x = gelu_exact((v.x - mu) * rs * g.x + bb.x);
    o.y = gelu_exact((v.y - mu) * rs * g.y + bb.y);
    o.z = gelu_exact((v.z - mu) * rs * g.z + bb.z);
    o.w = gelu_exact((v.w - mu) * rs * g.w + bb.w);
    *(float4*)(out + (size_t)(blockIdx.x * 16 + rr) * DIM + lane * 4) = o;
  }
}

// ---------------------------------------------------------------------------
extern "C" void kernel_launch(void* const* d_in, const int* in_sizes, int n_in,
                              void* d_out, int out_size, void* d_ws, size_t ws_size,
                              hipStream_t stream) {
  const float* x     = (const float*)d_in[0];
  const int*   ei    = (const int*)  d_in[1];
  const float* W_l   = (const float*)d_in[2];
  const float* b_l   = (const float*)d_in[3];
  const float* W_r   = (const float*)d_in[4];
  const float* gamma = (const float*)d_in[5];
  const float* beta  = (const float*)d_in[6];
  float* out = (float*)d_out;

  const int N = in_sizes[0] / DIM;     // 100000
  const int E = in_sizes[1] / 2;       // 3200000

  char* ws = (char*)d_ws;
  float* summed = (float*)ws;                                    // N*128 f32
  float* deg    = (float*)(ws + (size_t)N * DIM * sizeof(float));// N f32
  __hip_bfloat16* wl_swz =
      (__hip_bfloat16*)(ws + (size_t)N * DIM * sizeof(float) + (size_t)N * sizeof(float));
  __hip_bfloat16* wr_swz = wl_swz + DIM * DIM;

  // 1) zero accumulators (summed + deg), every call (determinism).
  {
    int n4 = (N * DIM + N) / 4;
    int blocks = (n4 + 255) / 256;
    zero_kernel<<<blocks, 256, 0, stream>>>((float4*)summed, n4);
  }
  // 2) swizzle weights into WMMA B layout (bf16).
  swizzle_w_kernel<<<(DIM * DIM + 255) / 256, 256, 0, stream>>>(W_l, wl_swz);
  swizzle_w_kernel<<<(DIM * DIM + 255) / 256, 256, 0, stream>>>(W_r, wr_swz);
  // 3) scatter-mean numerators: one wave per edge.
  {
    int blocks = (E + 7) / 8;  // E waves * 32 lanes / 256 threads
    scatter_add_kernel<<<blocks, 256, 0, stream>>>(x, ei, summed, deg, E);
  }
  // 4) fused dual-GEMM (WMMA bf16) + bias + LayerNorm + exact GELU.
  gemm_ln_gelu_kernel<<<N / 16, 256, 0, stream>>>(
      x, summed, deg, (const void*)wl_swz, (const void*)wr_swz,
      b_l, gamma, beta, out);
}
